// LinformerSelfAttention_CF_test_61014305407023
// MI455X (gfx1250) — matrix operand order repaired and checked by reading, not compiled, not run on hardware
//
#include <hip/hip_runtime.h>
#include <hip/hip_bf16.h>

// ---------------------------------------------------------------------------
// Linformer self-attention refresh for gfx1250 (MI455X), bf16 WMMA pipeline.
//   B=2, S=4096, D=1024, H=16, hd=64, R=2048, r_keep=1843
// ---------------------------------------------------------------------------

#define B_SZ   2
#define S_SZ   4096
#define D_SZ   1024
#define H_SZ   16
#define HD_SZ  64
#define R_SZ   2048
#define RKEEP  1843
#define M_ROWS (B_SZ * S_SZ)          // 8192

typedef __attribute__((ext_vector_type(16))) __bf16 v16bf;
typedef __attribute__((ext_vector_type(8)))  float  v8f;
typedef __attribute__((ext_vector_type(4)))  int    v4i;

#if defined(__has_builtin)
#if __has_builtin(__builtin_amdgcn_global_load_async_to_lds_b128) && \
    __has_builtin(__builtin_amdgcn_s_wait_asynccnt)
#define HAVE_ASYNC_LDS 1
#endif
#endif
#ifndef HAVE_ASYNC_LDS
#define HAVE_ASYNC_LDS 0
#endif

#if HAVE_ASYNC_LDS
typedef __attribute__((address_space(1))) v4i* g_v4i_p;   // global int4*
typedef __attribute__((address_space(3))) v4i* l_v4i_p;   // LDS int4*
#endif

// ---------------- bf16 helpers (stored as ushort everywhere) ---------------
__device__ inline unsigned short f_to_bfu(float f) {
    unsigned int x = __float_as_uint(f);
    unsigned int r = x + 0x7FFFu + ((x >> 16) & 1u);   // round-to-nearest-even
    return (unsigned short)(r >> 16);
}
__device__ inline __bf16 us_to_bf(unsigned short u) {
    return __builtin_bit_cast(__bf16, u);
}
__device__ inline v8f v8f_zero() {
    v8f z;
#pragma unroll
    for (int i = 0; i < 8; ++i) z[i] = 0.0f;
    return z;
}
__device__ inline unsigned int pack_bf2(float a, float b) {
    return (unsigned int)f_to_bfu(a) | ((unsigned int)f_to_bfu(b) << 16);
}

// ---------------- WMMA operand loaders (contiguous k = 16*(lane/16) + e, same in A and B) -----------------
// A operand: 16x32 bf16 (MxK), src row-major [m][k], leading dim ld.
__device__ inline v16bf load_a_tile(const unsigned short* s, int ld) {
    int lane = threadIdx.x & 31;
    int g = lane >> 4, m = lane & 15;
    v16bf a;
#pragma unroll
    for (int e = 0; e < 16; ++e) {
        int k = g * 16 + e;
        a[e] = us_to_bf(s[m * ld + k]);
    }
    return a;
}
// B operand: 32x16 (KxN), src stored [n][k] row-major (e.g. weight W[n][k]).
__device__ inline v16bf load_b_tile_nk(const unsigned short* s, int ldk) {
    int lane = threadIdx.x & 31;
    int g = lane >> 4, n = lane & 15;
    v16bf b;
#pragma unroll
    for (int e = 0; e < 16; ++e) {
        int k = g * 16 + e;
        b[e] = us_to_bf(s[n * ldk + k]);
    }
    return b;
}
// B operand: 32x16 (KxN), src stored [k][n] row-major (e.g. V tile [r][d]).
__device__ inline v16bf load_b_tile_kn(const unsigned short* s, int ldn) {
    int lane = threadIdx.x & 31;
    int g = lane >> 4, n = lane & 15;
    v16bf b;
#pragma unroll
    for (int e = 0; e < 16; ++e) {
        int k = g * 16 + e;
        b[e] = us_to_bf(s[k * ldn + n]);
    }
    return b;
}

__device__ inline float half_reduce_max(float v) {
#pragma unroll
    for (int m = 1; m < 16; m <<= 1) v = fmaxf(v, __shfl_xor(v, m, 32));
    return v;
}
__device__ inline float half_reduce_sum(float v) {
#pragma unroll
    for (int m = 1; m < 16; m <<= 1) v += __shfl_xor(v, m, 32);
    return v;
}

// ---- contiguous 8KB (64x64 bf16) global -> LDS tile copy, 128 threads -----
__device__ inline void tile_copy_8kb(unsigned short* __restrict__ dst_lds,
                                     const unsigned short* __restrict__ src,
                                     int tid)
{
#if HAVE_ASYNC_LDS
#pragma unroll
    for (int e = tid * 8; e < 64 * 64; e += 128 * 8) {
        __builtin_amdgcn_global_load_async_to_lds_b128(
            (g_v4i_p)(src + e), (l_v4i_p)(dst_lds + e), 0, 0);
    }
#else
#pragma unroll
    for (int e = tid * 8; e < 64 * 64; e += 128 * 8)
        *(uint4*)&dst_lds[e] = *(const uint4*)&src[e];
#endif
}
__device__ inline void tile_copy_wait()
{
#if HAVE_ASYNC_LDS
    __builtin_amdgcn_s_wait_asynccnt(0);
#endif
}

// ---------------------------------------------------------------------------
// GEMM: C[M=8192, N=1024] = A[8192,1024] @ W[1024,1024]^T + bias
// A_BF16=false: A f32, output bf16 in [B,H,S,hd] layout (Q/K/V projections).
// A_BF16=true : A bf16, output f32 row-major (final Wo projection).
// Block: 128 threads (4 waves), 64x64 C tile, k-chunks of 64.
// ---------------------------------------------------------------------------
template <bool A_BF16>
__global__ void __launch_bounds__(128)
gemm_wmma_kernel(const float* __restrict__ Af, const unsigned short* __restrict__ Abf,
                 const float* __restrict__ W, const float* __restrict__ bias,
                 float* __restrict__ Cf, unsigned short* __restrict__ Cbf)
{
    __shared__ unsigned short lA[64 * 64];
    __shared__ unsigned short lW[64 * 64];

    const int tid = threadIdx.x;
    const int w = tid >> 5;
    const int lane = tid & 31;
    const int g = lane >> 4, nlane = lane & 15;
    const int n0 = blockIdx.x * 64;
    const int m0 = blockIdx.y * 64;

    v8f acc[4];
#pragma unroll
    for (int i = 0; i < 4; ++i) acc[i] = v8f_zero();

    for (int kc = 0; kc < D_SZ / 64; ++kc) {
        const int k0 = kc * 64;
        // stage A (64x64) and W (64x64) tiles as bf16
#pragma unroll
        for (int e = tid * 4; e < 64 * 64; e += 128 * 4) {
            int r = e >> 6, c = e & 63;
            if constexpr (A_BF16) {
                *(uint2*)&lA[e] = *(const uint2*)&Abf[(long)(m0 + r) * D_SZ + k0 + c];
            } else {
                float4 f = *(const float4*)&Af[(long)(m0 + r) * D_SZ + k0 + c];
                uint2 v; v.x = pack_bf2(f.x, f.y); v.y = pack_bf2(f.z, f.w);
                *(uint2*)&lA[e] = v;
            }
            float4 fw = *(const float4*)&W[(long)(n0 + r) * D_SZ + k0 + c];
            uint2 vw; vw.x = pack_bf2(fw.x, fw.y); vw.y = pack_bf2(fw.z, fw.w);
            *(uint2*)&lW[e] = vw;
        }
        __syncthreads();

        v16bf a0 = load_a_tile(&lA[(w * 16) * 64 + 0], 64);
        v16bf a1 = load_a_tile(&lA[(w * 16) * 64 + 32], 64);
#pragma unroll
        for (int nt = 0; nt < 4; ++nt) {
            acc[nt] = __builtin_amdgcn_wmma_f32_16x16x32_bf16(
                false, a0, false, load_b_tile_nk(&lW[(nt * 16) * 64 + 0], 64),
                (short)0, acc[nt], false, false);
            acc[nt] = __builtin_amdgcn_wmma_f32_16x16x32_bf16(
                false, a1, false, load_b_tile_nk(&lW[(nt * 16) * 64 + 32], 64),
                (short)0, acc[nt], false, false);
        }
        __syncthreads();
    }

#pragma unroll
    for (int nt = 0; nt < 4; ++nt) {
        int n = n0 + nt * 16 + nlane;
        float bv = bias[n];
#pragma unroll
        for (int i = 0; i < 8; ++i) {
            int m = m0 + w * 16 + i + 8 * g;
            float val = acc[nt][i] + bv;
            if constexpr (!A_BF16) {   // head layout, bf16 out
                int b = m >> 12, s = m & (S_SZ - 1);
                int h = n >> 6, d = n & 63;
                Cbf[(((long)(b * H_SZ + h)) * S_SZ + s) * HD_SZ + d] = f_to_bfu(val);
            } else {                   // row-major f32 out
                Cf[(long)m * D_SZ + n] = val;
            }
        }
    }
}

// ---------------------------------------------------------------------------
// Gather Kg/Vg: [B,H,R,hd] <- K/V [B,H,S,hd] at topk_indices (32 thr / row)
// ---------------------------------------------------------------------------
__global__ void gather_kv_kernel(const unsigned int* __restrict__ K,
                                 const unsigned int* __restrict__ V,
                                 const int* __restrict__ idx,
                                 unsigned int* __restrict__ Kg,
                                 unsigned int* __restrict__ Vg)
{
    int r = blockIdx.x, h = blockIdx.y, b = blockIdx.z, t = threadIdx.x;
    int src = idx[(b * H_SZ + h) * R_SZ + r];
    long bi = (((long)(b * H_SZ + h)) * S_SZ + src) * (HD_SZ / 2);
    long bo = (((long)(b * H_SZ + h)) * R_SZ + r) * (HD_SZ / 2);
    Kg[bo + t] = K[bi + t];
    Vg[bo + t] = V[bi + t];
}

__global__ void zero_f32_kernel(float* p, int n)
{
    int i = blockIdx.x * 256 + threadIdx.x;
    if (i < n) p[i] = 0.0f;
}

// ---------------------------------------------------------------------------
// Attention: per (b,h, 64-query block). Two-pass flash softmax over R=2048.
// Pass 1: row max/sum. Pass 2: p = exp(s-m)/l, importance column sums, p@V.
// ---------------------------------------------------------------------------
__global__ void __launch_bounds__(128)
attn_flash_kernel(const unsigned short* __restrict__ Q,
                  const unsigned short* __restrict__ Kg,
                  const unsigned short* __restrict__ Vg,
                  unsigned short* __restrict__ Obf,
                  float* __restrict__ imp)
{
    __shared__ unsigned short lQ[64 * 64];
    __shared__ unsigned short lK[64 * 64];
    __shared__ unsigned short lV[64 * 64];
    __shared__ unsigned short lP[4][16 * 64];
    __shared__ float limp[R_SZ];

    const int tid = threadIdx.x;
    const int w = tid >> 5;
    const int lane = tid & 31;
    const int g = lane >> 4, nlane = lane & 15;
    const int q0 = blockIdx.x * 64;
    const int h = blockIdx.y;
    const int b = blockIdx.z;
    const long bh = b * H_SZ + h;
    const float scale = 0.125f;   // 1/sqrt(64)

    for (int i = tid; i < R_SZ; i += 128) limp[i] = 0.0f;

    tile_copy_8kb(lQ, Q + (bh * S_SZ + q0) * HD_SZ, tid);
    tile_copy_wait();
    __syncthreads();

    float mrow[8], lrow[8];
#pragma unroll
    for (int i = 0; i < 8; ++i) { mrow[i] = -1e30f; lrow[i] = 0.0f; }

    // ---------------- Pass 1: row max / row sum ----------------
    for (int kn = 0; kn < R_SZ / 64; ++kn) {
        tile_copy_8kb(lK, Kg + (bh * R_SZ + kn * 64) * HD_SZ, tid);
        tile_copy_wait();
        __syncthreads();

        v16bf a0 = load_a_tile(&lQ[(w * 16) * 64 + 0], 64);
        v16bf a1 = load_a_tile(&lQ[(w * 16) * 64 + 32], 64);
        v8f sc[4];
#pragma unroll
        for (int nt = 0; nt < 4; ++nt) {
            v8f s = v8f_zero();
            s = __builtin_amdgcn_wmma_f32_16x16x32_bf16(
                false, a0, false, load_b_tile_nk(&lK[(nt * 16) * 64 + 0], 64),
                (short)0, s, false, false);
            s = __builtin_amdgcn_wmma_f32_16x16x32_bf16(
                false, a1, false, load_b_tile_nk(&lK[(nt * 16) * 64 + 32], 64),
                (short)0, s, false, false);
#pragma unroll
            for (int i = 0; i < 8; ++i) s[i] *= scale;
            sc[nt] = s;
        }
#pragma unroll
        for (int i = 0; i < 8; ++i) {
            float tmax = fmaxf(fmaxf(sc[0][i], sc[1][i]), fmaxf(sc[2][i], sc[3][i]));
            tmax = half_reduce_max(tmax);
            float mn = fmaxf(mrow[i], tmax);
            float ps = __expf(sc[0][i] - mn) + __expf(sc[1][i] - mn) +
                       __expf(sc[2][i] - mn) + __expf(sc[3][i] - mn);
            ps = half_reduce_sum(ps);
            lrow[i] = lrow[i] * __expf(mrow[i] - mn) + ps;
            mrow[i] = mn;
        }
        __syncthreads();
    }

    float rinv[8];
#pragma unroll
    for (int i = 0; i < 8; ++i) rinv[i] = 1.0f / lrow[i];

    v8f oacc[4];
#pragma unroll
    for (int i = 0; i < 4; ++i) oacc[i] = v8f_zero();

    // ---------------- Pass 2: p, importance, p @ V ----------------
    for (int kn = 0; kn < R_SZ / 64; ++kn) {
        tile_copy_8kb(lK, Kg + (bh * R_SZ + kn * 64) * HD_SZ, tid);
        tile_copy_8kb(lV, Vg + (bh * R_SZ + kn * 64) * HD_SZ, tid);
        tile_copy_wait();
        __syncthreads();

        v16bf a0 = load_a_tile(&lQ[(w * 16) * 64 + 0], 64);
        v16bf a1 = load_a_tile(&lQ[(w * 16) * 64 + 32], 64);
        float cs[4] = {0.f, 0.f, 0.f, 0.f};
#pragma unroll
        for (int nt = 0; nt < 4; ++nt) {
            v8f s = v8f_zero();
            s = __builtin_amdgcn_wmma_f32_16x16x32_bf16(
                false, a0, false, load_b_tile_nk(&lK[(nt * 16) * 64 + 0], 64),
                (short)0, s, false, false);
            s = __builtin_amdgcn_wmma_f32_16x16x32_bf16(
                false, a1, false, load_b_tile_nk(&lK[(nt * 16) * 64 + 32], 64),
                (short)0, s, false, false);
#pragma unroll
            for (int i = 0; i < 8; ++i) {
                float p = __expf(s[i] * scale - mrow[i]) * rinv[i];
                cs[nt] += p;
                lP[w][(i + 8 * g) * 64 + nt * 16 + nlane] = f_to_bfu(p);
            }
        }
#pragma unroll
        for (int nt = 0; nt < 4; ++nt)
            atomicAdd(&limp[kn * 64 + nt * 16 + nlane], cs[nt]);

        // p (16x64) @ V tile (64x64) -> oacc (16x64)
        v16bf p0 = load_a_tile(&lP[w][0], 64);
        v16bf p1 = load_a_tile(&lP[w][32], 64);
#pragma unroll
        for (int nt = 0; nt < 4; ++nt) {
            oacc[nt] = __builtin_amdgcn_wmma_f32_16x16x32_bf16(
                false, p0, false, load_b_tile_kn(&lV[0 * 64 + nt * 16], 64),
                (short)0, oacc[nt], false, false);
            oacc[nt] = __builtin_amdgcn_wmma_f32_16x16x32_bf16(
                false, p1, false, load_b_tile_kn(&lV[32 * 64 + nt * 16], 64),
                (short)0, oacc[nt], false, false);
        }
        __syncthreads();
    }

    // write attention output bf16 into [B,S,D] layout (col = h*64 + d)
#pragma unroll
    for (int nt = 0; nt < 4; ++nt) {
        int col = h * HD_SZ + nt * 16 + nlane;
#pragma unroll
        for (int i = 0; i < 8; ++i) {
            int srow = q0 + w * 16 + i + 8 * g;
            Obf[((long)b * S_SZ + srow) * D_SZ + col] = f_to_bfu(oacc[nt][i]);
        }
    }
    __syncthreads();
    for (int i = tid; i < R_SZ; i += 128)
        atomicAdd(&imp[bh * R_SZ + i], limp[i]);
}

// ---------------------------------------------------------------------------
// Top-k: per (b,h), bitonic sort 2048 keys in LDS (desc value, asc index).
// ---------------------------------------------------------------------------
__global__ void __launch_bounds__(256)
topk_kernel(const float* __restrict__ imp, int* __restrict__ outIdx)
{
    __shared__ unsigned long long keys[R_SZ];
    const int tid = threadIdx.x;
    const int bh = blockIdx.x;

    for (int i = tid; i < R_SZ; i += 256) {
        unsigned int ub = __float_as_uint(imp[(long)bh * R_SZ + i]);
        ub = (ub & 0x80000000u) ? ~ub : (ub | 0x80000000u);  // monotonic asc
        keys[i] = ((unsigned long long)(~ub) << 32) | (unsigned int)i;
    }
    for (int k = 2; k <= R_SZ; k <<= 1) {
        for (int j = k >> 1; j > 0; j >>= 1) {
            __syncthreads();
            for (int i = tid; i < R_SZ; i += 256) {
                int l = i ^ j;
                if (l > i) {
                    bool up = ((i & k) == 0);
                    unsigned long long a = keys[i], c = keys[l];
                    if ((a > c) == up) { keys[i] = c; keys[l] = a; }
                }
            }
        }
    }
    __syncthreads();
    for (int i = tid; i < RKEEP; i += 256)
        outIdx[(long)bh * RKEEP + i] = (int)(keys[i] & 0xFFFFFFFFu);
}

// ---------------------------------------------------------------------------
extern "C" void kernel_launch(void* const* d_in, const int* in_sizes, int n_in,
                              void* d_out, int out_size, void* d_ws, size_t ws_size,
                              hipStream_t stream)
{
    const float* x  = (const float*)d_in[0];
    const float* Wq = (const float*)d_in[1];
    const float* bq = (const float*)d_in[2];
    const float* Wk = (const float*)d_in[3];
    const float* bk = (const float*)d_in[4];
    const float* Wv = (const float*)d_in[5];
    const float* bv = (const float*)d_in[6];
    const float* Wo = (const float*)d_in[7];
    const float* bo = (const float*)d_in[8];
    const int* tki  = (const int*)d_in[9];

    // workspace layout (bytes)
    const size_t SZ_PROJ = (size_t)M_ROWS * D_SZ * 2;              // 16 MB bf16
    const size_t SZ_KV   = (size_t)B_SZ * H_SZ * R_SZ * HD_SZ * 2; //  8 MB bf16
    char* ws = (char*)d_ws;
    unsigned short* Qb = (unsigned short*)(ws);
    unsigned short* Kb = (unsigned short*)(ws + SZ_PROJ);
    unsigned short* Vb = (unsigned short*)(ws + 2 * SZ_PROJ);
    unsigned short* Kg = (unsigned short*)(ws + 3 * SZ_PROJ);
    unsigned short* Vg = (unsigned short*)(ws + 3 * SZ_PROJ + SZ_KV);
    unsigned short* AO = (unsigned short*)(ws + 3 * SZ_PROJ + 2 * SZ_KV);
    float* imp         = (float*)(ws + 4 * SZ_PROJ + 2 * SZ_KV);

    float* outF = (float*)d_out;
    int* outIdx = (int*)(outF + (size_t)B_SZ * S_SZ * D_SZ);

    // 0) zero importance accumulator
    zero_f32_kernel<<<(B_SZ * H_SZ * R_SZ + 255) / 256, 256, 0, stream>>>(
        imp, B_SZ * H_SZ * R_SZ);

    // 1) Q/K/V projections -> bf16 [B,H,S,hd]
    dim3 gg(D_SZ / 64, M_ROWS / 64);
    gemm_wmma_kernel<false><<<gg, 128, 0, stream>>>(x, nullptr, Wq, bq, nullptr, Qb);
    gemm_wmma_kernel<false><<<gg, 128, 0, stream>>>(x, nullptr, Wk, bk, nullptr, Kb);
    gemm_wmma_kernel<false><<<gg, 128, 0, stream>>>(x, nullptr, Wv, bv, nullptr, Vb);

    // 2) gather reduced K/V (b32 lanes)
    gather_kv_kernel<<<dim3(R_SZ, H_SZ, B_SZ), 32, 0, stream>>>(
        (const unsigned int*)Kb, (const unsigned int*)Vb, tki,
        (unsigned int*)Kg, (unsigned int*)Vg);

    // 3) attention + importance sums
    attn_flash_kernel<<<dim3(S_SZ / 64, H_SZ, B_SZ), 128, 0, stream>>>(
        Qb, Kg, Vg, AO, imp);

    // 4) output projection -> f32 d_out
    gemm_wmma_kernel<true><<<gg, 128, 0, stream>>>(nullptr, AO, Wo, bo, outF, nullptr);

    // 5) top-k indices -> int tail of d_out
    topk_kernel<<<B_SZ * H_SZ, 256, 0, stream>>>(imp, outIdx);
}
